// NeuralODE_8985071583306
// MI455X (gfx1250) — matrix operand order repaired and checked by reading, not compiled, not run on hardware
//
#include <hip/hip_runtime.h>
#include <cstddef>
#include <cstdint>

// Problem constants (from reference)
#define HB  512   // batch
#define NT  64    // timesteps
#define DX  4     // per-step feature dim (time + 3 feats)
#define MT  4     // META
#define HH  512   // hidden
#define DIN 8     // D + META

typedef _Float16 half_t;
typedef __attribute__((ext_vector_type(16))) _Float16 v16h;
typedef __attribute__((ext_vector_type(8)))  float    v8f;
typedef unsigned int u32x4 __attribute__((ext_vector_type(4)));
typedef int          i32x4 __attribute__((ext_vector_type(4)));
typedef int          i32x8 __attribute__((ext_vector_type(8)));

enum { ACT_NONE = 0, ACT_RELU = 1, ACT_SELU = 2 };

// ---- CDNA5 Tensor Data Mover availability ----
#if defined(__has_builtin)
#  if __has_builtin(__builtin_amdgcn_tensor_load_to_lds)
#    define HAVE_TDM 1
#  endif
#  if __has_builtin(__builtin_amdgcn_s_wait_tensorcnt)
#    define WAIT_TENSOR() __builtin_amdgcn_s_wait_tensorcnt(0)
#  endif
#endif
#ifndef HAVE_TDM
#  define HAVE_TDM 0
#endif
#ifndef WAIT_TENSOR
#  define WAIT_TENSOR()
#endif

#if HAVE_TDM
// Issue a TDM 2-D tile load: rows x cols f16 elements, row stride
// `stride` elements, global -> LDS. Descriptor per ISA 8.3/8.4.
__device__ __forceinline__ void tdm_load_tile_f16(
    const half_t* gptr, void* lptr, int rows, int cols, int stride)
{
  const unsigned long long ga = (unsigned long long)(uintptr_t)gptr;
  const unsigned int lds = (unsigned int)(uintptr_t)lptr; // low 32b = LDS offset

  u32x4 g0;
  g0[0] = 1u;                                                 // count=1
  g0[1] = lds;                                                // lds_addr
  g0[2] = (unsigned int)(ga & 0xFFFFFFFFu);                   // global_addr[31:0]
  g0[3] = (unsigned int)((ga >> 32) & 0x01FFFFFFu) | 0x80000000u; // [56:32]|type=2

  const unsigned int dim0 = 0x40000000u;  // huge tensor dims: no OOB clipping
  const unsigned int dim1 = 0x40000000u;
  i32x8 g1;
  g1[0] = (int)(1u << 16);                                    // data_size=2B
  g1[1] = (int)((dim0 & 0xFFFFu) << 16);                      // tensor_dim0[15:0]
  g1[2] = (int)((dim0 >> 16) | ((dim1 & 0xFFFFu) << 16));     // d0[31:16], d1[15:0]
  g1[3] = (int)((dim1 >> 16) | ((unsigned)cols << 16));       // d1[31:16], tile_dim0
  g1[4] = (int)(unsigned)rows;                                // tile_dim1, tile_dim2=0
  g1[5] = (int)(unsigned)stride;                              // tensor_dim0_stride[31:0]
  g1[6] = 0;
  g1[7] = 0;

  i32x4 gz = {0, 0, 0, 0};
#  if __has_include(<hip/amd_detail/amd_gfx1250_TDM.h>)
  i32x8 gz8 = {0, 0, 0, 0, 0, 0, 0, 0};
  __builtin_amdgcn_tensor_load_to_lds(g0, g1, gz, gz, gz8, 0);
#  else
  __builtin_amdgcn_tensor_load_to_lds(g0, g1, gz, gz, 0);
#  endif
}
#endif

// ---------------------------------------------------------------------------
// WMMA GEMM:  O[M x Ntot] = act( X[M x K] * W[Ntot x K]^T + bias[Ntot] )
//   - 256 threads = 8 wave32; workgroup tile 64(M) x 64(N); each wave owns a
//     16x32 strip: one A fragment reused across two v_wmma_f32_16x16x32_f16.
//   - Double-buffered LDS (2 x (4KB A + 4KB B)); one barrier per K-chunk.
//   - B (weight) tiles staged by the Tensor Data Mover from wave 0
//     (tensor_load_to_lds + s_wait_tensorcnt), A tiles by vector loads.
//   - COMPOSE: A built on the fly as f16( z + c*dt(row)*kprev ), with
//     dt(row) = x[row, ti+1, 0] - x[row, ti, 0]  (RK4 stage-input fusion).
// ---------------------------------------------------------------------------
template <int ACT, bool COMPOSE, bool WF32, bool WF16>
__global__ __launch_bounds__(256) void gemm_wmma(
    const half_t* __restrict__ Xh,     // !COMPOSE: f16 input [M x K]
    const float*  __restrict__ Za,     // COMPOSE: z
    const float*  __restrict__ Kp,     // COMPOSE: k_prev
    const float*  __restrict__ xt,     // COMPOSE: raw x tensor (for dt)
    float cscale, int ti,
    const half_t* __restrict__ Wh,     // f16 weights [Ntot x K]
    const float*  __restrict__ bias,   // [Ntot]
    float* __restrict__ Of32, half_t* __restrict__ Of16,
    int K, int Ntot)
{
  __shared__ __align__(32) half_t As[2][64 * 32];
  __shared__ __align__(32) half_t Bs[2][64 * 32];   // [n][k]

  const int tid  = threadIdx.x;
  const int lane = tid & 31;
  const int wid  = tid >> 5;
  const int wm   = wid & 3;          // 0..3 -> m sub-tile
  const int wn   = wid >> 2;         // 0..1 -> 32-wide n strip
  const int tile_m = blockIdx.x * 64;
  const int tile_n = blockIdx.y * 64;

  v8f acc0 = {};
  v8f acc1 = {};

  // staging geometry: 64x32 halves = 2048 -> 8 halves / thread
  const int sidx = tid * 8;
  const int srow = sidx >> 5;
  const int scol = sidx & 31;
  const int gm   = tile_m + srow;

  float cd = 0.f;
  if (COMPOSE) {
    const float t0 = xt[((size_t)gm * NT + ti    ) * DX];
    const float t1 = xt[((size_t)gm * NT + ti + 1) * DX];
    cd = cscale * (t1 - t0);
  }

  auto stageA = [&](int buf, int kc) {
    if (COMPOSE) {
      const float* zp = Za + (size_t)gm * K + kc + scol;
      const float* kp = Kp + (size_t)gm * K + kc + scol;
      #pragma unroll
      for (int q = 0; q < 8; ++q)
        As[buf][sidx + q] = (half_t)(zp[q] + cd * kp[q]);
    } else {
      *(float4*)&As[buf][sidx] =
          *(const float4*)(Xh + (size_t)gm * K + kc + scol);
      __builtin_prefetch(Xh + (size_t)gm * K + kc + 32 + scol, 0, 1);
    }
  };
#if !HAVE_TDM
  auto stageB = [&](int buf, int kc) {
    *(float4*)&Bs[buf][sidx] =
        *(const float4*)(Wh + (size_t)(tile_n + srow) * K + kc + scol);
  };
#endif

  // ---- preload chunk 0 into buffer 0 ----
  stageA(0, 0);
#if HAVE_TDM
  if (wid == 0) {
    tdm_load_tile_f16(Wh + (size_t)tile_n * K, &Bs[0][0], 64, 32, K);
    WAIT_TENSOR();
  }
#else
  stageB(0, 0);
#endif
  __syncthreads();

  int cur = 0;
  for (int kc = 0; kc < K; kc += 32) {
    const int nxt = cur ^ 1;
    const bool more = (kc + 32 < K);
    if (more) {
      stageA(nxt, kc + 32);
#if HAVE_TDM
      if (wid == 0)
        tdm_load_tile_f16(Wh + (size_t)tile_n * K + kc + 32, &Bs[nxt][0],
                          64, 32, K);
#else
      stageB(nxt, kc + 32);
#endif
    }

    // ---- A fragment (16x32 f16):
    // lane L: row = L&15, kbase = (L>>4)*16; halves e -> kbase+e (contiguous)
    const int arow = wm * 16 + (lane & 15);
    const int kb   = (lane >> 4) * 16;
    const half_t* ap = &As[cur][arow * 32 + kb];
    v16h afrag = *(const v16h*)ap;

    // ---- B fragments (32x16 f16): lane L: col = L&15,
    // K = (L<16 ? 0..15 : 16..31) contiguous; two n sub-tiles share A
    const int bcol = wn * 32 + (lane & 15);
    const int koff = (lane >> 4) * 16;
    v16h bfrag0 = *(const v16h*)&Bs[cur][bcol * 32 + koff];
    v16h bfrag1 = *(const v16h*)&Bs[cur][(bcol + 16) * 32 + koff];

    acc0 = __builtin_amdgcn_wmma_f32_16x16x32_f16(false, afrag, false, bfrag0,
                                                  (short)0, acc0, false, false);
    acc1 = __builtin_amdgcn_wmma_f32_16x16x32_f16(false, afrag, false, bfrag1,
                                                  (short)0, acc1, false, false);

#if HAVE_TDM
    if (wid == 0 && more) WAIT_TENSOR();
#endif
    __syncthreads();
    cur = nxt;
  }

  // ---- epilogue: bias + activation + store (C/D f32 layout: VGPR v holds
  // row m = v + 8*(lane>>4), col n = lane&15) ----
  const int m0 = tile_m + wm * 16 + 8 * (lane >> 4);
  #pragma unroll
  for (int h = 0; h < 2; ++h) {
    const v8f acc = h ? acc1 : acc0;
    const int n = tile_n + wn * 32 + 16 * h + (lane & 15);
    const float bv = bias[n];
    #pragma unroll
    for (int v = 0; v < 8; ++v) {
      float val = acc[v] + bv;
      if (ACT == ACT_RELU) val = fmaxf(val, 0.f);
      if (ACT == ACT_SELU) {
        const float alpha = 1.6732632423543772f, lam = 1.0507009873554805f;
        val = lam * (val > 0.f ? val : alpha * (__expf(val) - 1.f));
      }
      const size_t o = (size_t)(m0 + v) * Ntot + n;
      if (WF32) Of32[o] = val;
      if (WF16) Of16[o] = (half_t)val;
    }
  }
}

// ---------------------------------------------------------------------------
// GRU gate fusion: consumes G = h @ Whh^T + bhh (from WMMA GEMM), does the
// tiny K=8 input projection inline, updates h (f32 + f16 copy) in place.
// ---------------------------------------------------------------------------
__global__ __launch_bounds__(256) void gru_gate(
    const float* __restrict__ G,
    const float* __restrict__ x, const float* __restrict__ meta,
    const float* __restrict__ Wih, const float* __restrict__ bih,
    float* __restrict__ h, half_t* __restrict__ h16, int t)
{
  const int idx = blockIdx.x * 256 + threadIdx.x;
  const int b = idx >> 9;        // / HH
  const int j = idx & (HH - 1);

  float xm[DIN];
  const float* xb = x + ((size_t)b * NT + t) * DX;
  #pragma unroll
  for (int q = 0; q < DX; ++q) xm[q] = xb[q];
  #pragma unroll
  for (int q = 0; q < MT; ++q) xm[DX + q] = meta[b * MT + q];

  float gi_r = bih[j], gi_z = bih[HH + j], gi_n = bih[2 * HH + j];
  const float* wr = Wih + (size_t)j * DIN;
  const float* wz = Wih + (size_t)(HH + j) * DIN;
  const float* wn = Wih + (size_t)(2 * HH + j) * DIN;
  #pragma unroll
  for (int q = 0; q < DIN; ++q) {
    gi_r += xm[q] * wr[q];
    gi_z += xm[q] * wz[q];
    gi_n += xm[q] * wn[q];
  }

  const float* gb = G + (size_t)b * (3 * HH);   // bhh already folded in
  const float r  = 1.f / (1.f + __expf(-(gi_r + gb[j])));
  const float z  = 1.f / (1.f + __expf(-(gi_z + gb[HH + j])));
  const float nn = tanhf(gi_n + r * gb[2 * HH + j]);

  const float hv = h[idx];
  const float hnew = (1.f - z) * nn + z * hv;
  h[idx]   = hnew;
  h16[idx] = (half_t)hnew;
}

// z0 = eps * std + mean  (mean = e2[:, :H], std = e2[:, H:])
__global__ __launch_bounds__(256) void z0_kernel(
    const float* __restrict__ e2, const float* __restrict__ eps,
    float* __restrict__ z)
{
  const int idx = blockIdx.x * 256 + threadIdx.x;
  const int b = idx >> 9;
  const int j = idx & (HH - 1);
  const float mean = e2[(size_t)b * (2 * HH) + j];
  const float stdv = e2[(size_t)b * (2 * HH) + HH + j];
  z[idx] = eps[idx] * stdv + mean;
}

// RK4 combine: z += dt/6 * (k0 + 2k1 + 2k2 + k3), dt per batch row from x
__global__ __launch_bounds__(256) void rk4_update(
    float* __restrict__ z,
    const float* __restrict__ k0, const float* __restrict__ k1,
    const float* __restrict__ k2, const float* __restrict__ k3,
    const float* __restrict__ x, int ti)
{
  const int idx = blockIdx.x * 256 + threadIdx.x;
  const int b = idx >> 9;
  const float dt = x[((size_t)b * NT + ti + 1) * DX] -
                   x[((size_t)b * NT + ti    ) * DX];
  z[idx] += dt * (1.f / 6.f) *
            (k0[idx] + 2.f * k1[idx] + 2.f * k2[idx] + k3[idx]);
}

__global__ __launch_bounds__(256) void f32_to_f16(
    const float* __restrict__ in, half_t* __restrict__ out, int n)
{
  const int idx = blockIdx.x * 256 + threadIdx.x;
  if (idx < n) out[idx] = (half_t)in[idx];
}

__global__ __launch_bounds__(256) void init_h(
    float* __restrict__ h, half_t* __restrict__ h16)
{
  const int idx = blockIdx.x * 256 + threadIdx.x;
  h[idx]   = 0.f;
  h16[idx] = (half_t)0.f;
}

// out[b] = fc_W[:H] . z[b] + fc_W[H:] . meta[b] + fc_b
__global__ __launch_bounds__(256) void final_fc(
    const float* __restrict__ z, const float* __restrict__ meta,
    const float* __restrict__ fcW, const float* __restrict__ fcb,
    float* __restrict__ out)
{
  __shared__ float red[256];
  const int b = blockIdx.x;
  float s = 0.f;
  for (int j = threadIdx.x; j < HH; j += 256)
    s += z[(size_t)b * HH + j] * fcW[j];
  red[threadIdx.x] = s;
  __syncthreads();
  for (int st = 128; st > 0; st >>= 1) {
    if (threadIdx.x < st) red[threadIdx.x] += red[threadIdx.x + st];
    __syncthreads();
  }
  if (threadIdx.x == 0) {
    float v = red[0] + fcb[0];
    #pragma unroll
    for (int m = 0; m < MT; ++m) v += meta[b * MT + m] * fcW[HH + m];
    out[b] = v;
  }
}

// ---------------------------------------------------------------------------
extern "C" void kernel_launch(void* const* d_in, const int* in_sizes, int n_in,
                              void* d_out, int out_size, void* d_ws, size_t ws_size,
                              hipStream_t stream)
{
  const float* x    = (const float*)d_in[0];
  const float* meta = (const float*)d_in[1];
  const float* eps  = (const float*)d_in[2];
  const float* Wih  = (const float*)d_in[3];
  const float* Whh  = (const float*)d_in[4];
  const float* bih  = (const float*)d_in[5];
  const float* bhh  = (const float*)d_in[6];
  const float* eW1  = (const float*)d_in[7];
  const float* eb1  = (const float*)d_in[8];
  const float* eW2  = (const float*)d_in[9];
  const float* eb2  = (const float*)d_in[10];
  const float* oW1  = (const float*)d_in[11];
  const float* ob1  = (const float*)d_in[12];
  const float* oW2  = (const float*)d_in[13];
  const float* ob2  = (const float*)d_in[14];
  const float* fcW  = (const float*)d_in[15];
  const float* fcb  = (const float*)d_in[16];
  float* out = (float*)d_out;

  // ---- workspace carve-up (~17.5 MB) ----
  char* ws = (char*)d_ws;
  size_t off = 0;
  auto carve = [&](size_t bytes) -> char* {
    char* p = ws + off;
    off += (bytes + 255) & ~(size_t)255;
    return p;
  };
  half_t* Whh16 = (half_t*)carve((size_t)3 * HH * HH * sizeof(half_t));
  half_t* eW116 = (half_t*)carve((size_t)HH * HH * sizeof(half_t));
  half_t* eW216 = (half_t*)carve((size_t)2 * HH * HH * sizeof(half_t));
  half_t* oW116 = (half_t*)carve((size_t)HH * HH * sizeof(half_t));
  half_t* oW216 = (half_t*)carve((size_t)HH * HH * sizeof(half_t));
  float*  hbuf  = (float*) carve((size_t)HB * HH * sizeof(float));
  half_t* h16   = (half_t*)carve((size_t)HB * HH * sizeof(half_t));
  float*  G     = (float*) carve((size_t)HB * 3 * HH * sizeof(float));
  half_t* e116  = (half_t*)carve((size_t)HB * HH * sizeof(half_t));
  float*  e2    = (float*) carve((size_t)HB * 2 * HH * sizeof(float));
  float*  z     = (float*) carve((size_t)HB * HH * sizeof(float));
  float*  k0    = (float*) carve((size_t)HB * HH * sizeof(float));
  float*  k1    = (float*) carve((size_t)HB * HH * sizeof(float));
  float*  k2    = (float*) carve((size_t)HB * HH * sizeof(float));
  float*  k3    = (float*) carve((size_t)HB * HH * sizeof(float));
  half_t* a16   = (half_t*)carve((size_t)HB * HH * sizeof(half_t));
  (void)ws_size; (void)n_in; (void)in_sizes; (void)out_size;

  const int EW = HB * HH / 256;   // elementwise grid

  // ---- convert weights to f16 once per launch (deterministic) ----
  f32_to_f16<<<(3 * HH * HH + 255) / 256, 256, 0, stream>>>(Whh, Whh16, 3 * HH * HH);
  f32_to_f16<<<(HH * HH + 255) / 256, 256, 0, stream>>>(eW1, eW116, HH * HH);
  f32_to_f16<<<(2 * HH * HH + 255) / 256, 256, 0, stream>>>(eW2, eW216, 2 * HH * HH);
  f32_to_f16<<<(HH * HH + 255) / 256, 256, 0, stream>>>(oW1, oW116, HH * HH);
  f32_to_f16<<<(HH * HH + 255) / 256, 256, 0, stream>>>(oW2, oW216, HH * HH);
  init_h<<<EW, 256, 0, stream>>>(hbuf, h16);

  // ---- GRU over 64 steps: WMMA GEMM (bhh fused as bias) + gate fusion ----
  dim3 gGRU(HB / 64, (3 * HH) / 64);
  for (int t = 0; t < NT; ++t) {
    gemm_wmma<ACT_NONE, false, true, false><<<gGRU, 256, 0, stream>>>(
        h16, nullptr, nullptr, nullptr, 0.f, 0, Whh16, bhh, G, nullptr, HH, 3 * HH);
    gru_gate<<<EW, 256, 0, stream>>>(G, x, meta, Wih, bih, hbuf, h16, t);
  }

  // ---- encoder: relu(h@W1^T+b1) @ W2^T + b2, then reparameterize ----
  gemm_wmma<ACT_RELU, false, false, true><<<dim3(HB / 64, HH / 64), 256, 0, stream>>>(
      h16, nullptr, nullptr, nullptr, 0.f, 0, eW116, eb1, nullptr, e116, HH, HH);
  gemm_wmma<ACT_NONE, false, true, false><<<dim3(HB / 64, (2 * HH) / 64), 256, 0, stream>>>(
      e116, nullptr, nullptr, nullptr, 0.f, 0, eW216, eb2, e2, nullptr, HH, 2 * HH);
  z0_kernel<<<EW, 256, 0, stream>>>(e2, eps, z);

  // ---- latent ODE: batched fixed-step RK4 across the 63 observation
  // intervals; drift = selu(z@W1^T+b1)@W2^T+b2. Stage input z + c*dt*k_prev
  // is fused into the first GEMM's A staging (COMPOSE). ----
  const float cs[4] = {0.f, 0.5f, 0.5f, 1.f};
  float* karr[4] = {k0, k1, k2, k3};
  dim3 gODE(HB / 64, HH / 64);
  for (int i = 0; i < NT - 1; ++i) {
    for (int s = 0; s < 4; ++s) {
      const float* kprev = (s == 0) ? z : karr[s - 1];
      gemm_wmma<ACT_SELU, true, false, true><<<gODE, 256, 0, stream>>>(
          nullptr, z, kprev, x, cs[s], i, oW116, ob1, nullptr, a16, HH, HH);
      gemm_wmma<ACT_NONE, false, true, false><<<gODE, 256, 0, stream>>>(
          a16, nullptr, nullptr, nullptr, 0.f, 0, oW216, ob2, karr[s], nullptr, HH, HH);
    }
    rk4_update<<<EW, 256, 0, stream>>>(z, k0, k1, k2, k3, x, i);
  }

  // ---- final FC ----
  final_fc<<<HB, 256, 0, stream>>>(z, meta, fcW, fcb, out);
}
